// SelfAttentionLayer_50989851738138
// MI455X (gfx1250) — matrix operand order repaired and checked by reading, not compiled, not run on hardware
//
#include <hip/hip_runtime.h>

// ---------------------------------------------------------------------------
// Self-attention (B=8, N=2048, D=1024) for gfx1250 / MI455X.
// GEMMs: v_wmma_f32_16x16x32_bf16, double-buffered LDS staging fed by
// global_load_async_to_lds_b128 (ASYNCcnt), fp32 accumulate.
// ---------------------------------------------------------------------------

typedef __bf16          v16bf __attribute__((ext_vector_type(16)));
typedef float           v8f   __attribute__((ext_vector_type(8)));
typedef unsigned short  us8   __attribute__((ext_vector_type(8)));

#define DEV __device__ __forceinline__

constexpr int BATCH = 8;
constexpr int SEQ   = 2048;
constexpr int DIM   = 1024;

constexpr int LDSPITCH = 40;   // 32 K-elems padded to 40 (80B = 20 banks stride)

DEV unsigned short f2bf(float f) {
    unsigned u = __builtin_bit_cast(unsigned, f);
    unsigned r = u + 0x7FFFu + ((u >> 16) & 1u);   // round-to-nearest-even
    return (unsigned short)(r >> 16);
}
DEV float bf2f(unsigned short h) {
    return __builtin_bit_cast(float, (unsigned)h << 16);
}

// 16B global -> LDS async copy (ASYNCcnt-tracked, no VGPR round trip)
DEV void async_cp16(unsigned lds_byte_addr, const void* gptr) {
    asm volatile("global_load_async_to_lds_b128 %0, %1, off"
                 :: "v"(lds_byte_addr), "v"(gptr)
                 : "memory");
}
DEV void wait_async0() {
    asm volatile("s_wait_asynccnt 0x0" ::: "memory");
}

// --------------------------- fp32 -> bf16 convert ---------------------------
__global__ __launch_bounds__(256) void cvt_f32_bf16(const float* __restrict__ in,
                                                    unsigned short* __restrict__ out,
                                                    int n8) {
    int i = blockIdx.x * blockDim.x + threadIdx.x;
    if (i >= n8) return;
    float4 a = ((const float4*)in)[i * 2 + 0];
    float4 b = ((const float4*)in)[i * 2 + 1];
    us8 o;
    o[0] = f2bf(a.x); o[1] = f2bf(a.y); o[2] = f2bf(a.z); o[3] = f2bf(a.w);
    o[4] = f2bf(b.x); o[5] = f2bf(b.y); o[6] = f2bf(b.z); o[7] = f2bf(b.w);
    ((us8*)out)[i] = o;
}

// ------------------------------- WMMA GEMM ---------------------------------
// C[M,Nc] = A[M,K] * Bt[Nc,K]^T  (+ bias[col]), batched over grid.z.
// Block = 8 waves; wave computes a 64x64 C tile; block covers 128 x 256.
// K staged through LDS in 32-deep double-buffered slabs via async copies.
// MODE 0: bf16 row-major output (bias optional)        -> Q/K proj, scores
// MODE 1: bf16 per-batch transposed output [b][e][m]   -> V proj
// MODE 2: fp32 row-major output                        -> final attention out
template<int MODE>
__global__ __launch_bounds__(256) void wmma_gemm(
    const unsigned short* __restrict__ A,  size_t sAz, int lda,
    const unsigned short* __restrict__ Bt, size_t sBz, int ldb,
    void* __restrict__ Cv,                 size_t sCz, int ldc,
    const float* __restrict__ bias, int K)
{
    __shared__ unsigned short As[2][128 * LDSPITCH];   // 20 KB
    __shared__ unsigned short BsS[2][256 * LDSPITCH];  // 40 KB

    const int tid  = threadIdx.x;
    const int lane = tid & 31;
    const int wave = tid >> 5;
    const int wm   = wave >> 2;                    // 0..1
    const int wn   = wave & 3;                     // 0..3
    const int mblk = blockIdx.y * 128;
    const int nblk = blockIdx.x * 256;
    const int z    = blockIdx.z;

    const unsigned short* Az = A  + (size_t)z * sAz;
    const unsigned short* Bz = Bt + (size_t)z * sBz;

    // ---- async-copy work distribution: 1536 x 16B chunks, 6 per thread ----
    // A slab: 128 rows x 4 chunks; B slab: 256 rows x 4 chunks.
    const int ar0 = tid >> 2;
    const int ar1 = (tid + 256) >> 2;
    const int ac  = (tid & 3) * 8;                 // K-chunk offset (elements)
    const unsigned short* gA0 = Az + (size_t)(mblk + ar0) * lda + ac;
    const unsigned short* gA1 = Az + (size_t)(mblk + ar1) * lda + ac;
    const unsigned lA0 = (unsigned)(ar0 * LDSPITCH + ac) * 2u;
    const unsigned lA1 = (unsigned)(ar1 * LDSPITCH + ac) * 2u;

    const unsigned short* gB[4];
    unsigned lB[4];
    #pragma unroll
    for (int c2 = 0; c2 < 4; ++c2) {
        const int t = tid + 256 * c2;
        const int r = t >> 2;
        gB[c2] = Bz + (size_t)(nblk + r) * ldb + ac;
        lB[c2] = (unsigned)(r * LDSPITCH + ac) * 2u;
    }

    const unsigned aBase[2] = { (unsigned)(size_t)&As[0][0],
                                (unsigned)(size_t)&As[1][0] };
    const unsigned bBase[2] = { (unsigned)(size_t)&BsS[0][0],
                                (unsigned)(size_t)&BsS[1][0] };

    // prologue: stage K-slab 0 into buffer 0
    async_cp16(aBase[0] + lA0, gA0);
    async_cp16(aBase[0] + lA1, gA1);
    #pragma unroll
    for (int c2 = 0; c2 < 4; ++c2) async_cp16(bBase[0] + lB[c2], gB[c2]);
    wait_async0();
    __syncthreads();

    const int mr = lane & 15;          // row (A) / col (B) within 16-tile
    const int ua = (lane >> 4) << 3;   // A frag: lanes16-31 hold K=8..15,24..31
    const int ub = (lane >> 4) << 3;   // B frag: lanes16-31 hold K=8..15,24..31

    union Frag { v16bf v; us8 h[2]; };
    v8f c[4][4] = {};                  // 16 accumulator tiles

    for (int k = 0; k < K; k += 32) {
        const int buf = (k >> 5) & 1;

        // kick off async staging of the next K-slab into the other buffer
        if (k + 32 < K) {
            const int nb = buf ^ 1;
            const int go = k + 32;
            async_cp16(aBase[nb] + lA0, gA0 + go);
            async_cp16(aBase[nb] + lA1, gA1 + go);
            #pragma unroll
            for (int c2 = 0; c2 < 4; ++c2)
                async_cp16(bBase[nb] + lB[c2], gB[c2] + go);
        }

        // fragments from LDS (ds_load_b128 pairs), then 16 WMMAs
        Frag a[4];
        #pragma unroll
        for (int i = 0; i < 4; ++i) {
            const unsigned short* p =
                &As[buf][(wm * 64 + i * 16 + mr) * LDSPITCH + ua];
            a[i].h[0] = *(const us8*)(p);        // K = ua..ua+7
            a[i].h[1] = *(const us8*)(p + 16);   // K = 16+ua..23+ua
        }
        #pragma unroll
        for (int j = 0; j < 4; ++j) {
            Frag b;
            const unsigned short* p =
                &BsS[buf][(wn * 64 + j * 16 + mr) * LDSPITCH + ub];
            b.h[0] = *(const us8*)(p);           // K = ub..ub+7
            b.h[1] = *(const us8*)(p + 16);      // K = 16+ub..23+ub
            #pragma unroll
            for (int i = 0; i < 4; ++i) {
                c[i][j] = __builtin_amdgcn_wmma_f32_16x16x32_bf16(
                    false, a[i].v, false, b.v, (short)0, c[i][j], false, false);
            }
        }

        // my async copies done + everyone finished reading before overwrite
        wait_async0();
        __syncthreads();
    }

    // Epilogue. C tile layout: VGPR r -> M = r + (lane<16 ? 0 : 8), N = lane&15
    const int hi = (lane >> 4) << 3;
    #pragma unroll
    for (int j = 0; j < 4; ++j) {
        const int col = nblk + wn * 64 + j * 16 + mr;
        float bv = 0.f;
        if (MODE != 2 && bias != nullptr) bv = bias[col];
        #pragma unroll
        for (int i = 0; i < 4; ++i) {
            #pragma unroll
            for (int r = 0; r < 8; ++r) {
                const int row = mblk + wm * 64 + i * 16 + hi + r;
                const float val = c[i][j][r] + bv;
                if constexpr (MODE == 0) {
                    unsigned short* Cp = (unsigned short*)Cv + (size_t)z * sCz;
                    Cp[(size_t)row * ldc + col] = f2bf(val);
                } else if constexpr (MODE == 1) {
                    const int b  = row >> 11;          // row / SEQ
                    const int nn = row & (SEQ - 1);
                    unsigned short* Cp = (unsigned short*)Cv;
                    Cp[((size_t)b * DIM + col) * SEQ + nn] = f2bf(val);
                } else {
                    float* Cp = (float*)Cv + (size_t)z * sCz;
                    Cp[(size_t)row * ldc + col] = val;
                }
            }
        }
    }
}

// ------------------------------ row softmax --------------------------------
__global__ __launch_bounds__(256) void softmax_rows(unsigned short* __restrict__ s)
{
    const size_t row = blockIdx.x;
    unsigned short* p = s + row * (size_t)SEQ;
    const int t = threadIdx.x;

    __shared__ float smax[8];
    __shared__ float ssum[8];

    us8 hv = ((const us8*)p)[t];
    float v[8], mx = -3.4e38f;
    #pragma unroll
    for (int i = 0; i < 8; ++i) { v[i] = bf2f(hv[i]); mx = fmaxf(mx, v[i]); }

    #pragma unroll
    for (int off = 16; off; off >>= 1) mx = fmaxf(mx, __shfl_xor(mx, off, 32));
    if ((t & 31) == 0) smax[t >> 5] = mx;
    __syncthreads();
    float m = smax[0];
    #pragma unroll
    for (int w = 1; w < 8; ++w) m = fmaxf(m, smax[w]);

    float e[8], sum = 0.f;
    #pragma unroll
    for (int i = 0; i < 8; ++i) { e[i] = __expf(v[i] - m); sum += e[i]; }
    #pragma unroll
    for (int off = 16; off; off >>= 1) sum += __shfl_xor(sum, off, 32);
    if ((t & 31) == 0) ssum[t >> 5] = sum;
    __syncthreads();
    float tot = 0.f;
    #pragma unroll
    for (int w = 0; w < 8; ++w) tot += ssum[w];
    const float inv = 1.0f / tot;

    us8 o;
    #pragma unroll
    for (int i = 0; i < 8; ++i) o[i] = f2bf(e[i] * inv);
    ((us8*)p)[t] = o;
}

// ------------------------------ launch glue --------------------------------
extern "C" void kernel_launch(void* const* d_in, const int* in_sizes, int n_in,
                              void* d_out, int out_size, void* d_ws, size_t ws_size,
                              hipStream_t stream) {
    (void)in_sizes; (void)n_in; (void)out_size; (void)ws_size;

    const float* x  = (const float*)d_in[0];
    const float* Wq = (const float*)d_in[1];
    const float* bq = (const float*)d_in[2];
    const float* Wk = (const float*)d_in[3];
    const float* bk = (const float*)d_in[4];
    const float* Wv = (const float*)d_in[5];
    const float* bv = (const float*)d_in[6];

    constexpr size_t XE = (size_t)BATCH * SEQ * DIM;   // 16,777,216
    constexpr size_t WE = (size_t)DIM * DIM;           //  1,048,576
    constexpr size_t SE = (size_t)BATCH * SEQ * SEQ;   // 33,554,432

    unsigned short* ws  = (unsigned short*)d_ws;
    unsigned short* xb  = ws;             // x  bf16
    unsigned short* qb  = xb  + XE;       // q  bf16 [B*N, D]
    unsigned short* kb  = qb  + XE;       // k  bf16 [B*N, D]
    unsigned short* vtb = kb  + XE;       // v  bf16 transposed [B][D][N]
    unsigned short* sc  = vtb + XE;       // scores/probs bf16 [B][N][N]
    unsigned short* wqb = sc  + SE;
    unsigned short* wkb = wqb + WE;
    unsigned short* wvb = wkb + WE;

    // 1) converts
    cvt_f32_bf16<<<(int)(XE / 8 / 256), 256, 0, stream>>>(x,  xb,  (int)(XE / 8));
    cvt_f32_bf16<<<(int)(WE / 8 / 256), 256, 0, stream>>>(Wq, wqb, (int)(WE / 8));
    cvt_f32_bf16<<<(int)(WE / 8 / 256), 256, 0, stream>>>(Wk, wkb, (int)(WE / 8));
    cvt_f32_bf16<<<(int)(WE / 8 / 256), 256, 0, stream>>>(Wv, wvb, (int)(WE / 8));

    // 2) projections: M = B*N = 16384 rows, Nc = D = 1024, K = D
    dim3 gp(DIM / 256, (BATCH * SEQ) / 128, 1);
    wmma_gemm<0><<<gp, 256, 0, stream>>>(xb, 0, DIM, wqb, 0, DIM, qb,  0, DIM, bq, DIM);
    wmma_gemm<0><<<gp, 256, 0, stream>>>(xb, 0, DIM, wkb, 0, DIM, kb,  0, DIM, bk, DIM);
    wmma_gemm<1><<<gp, 256, 0, stream>>>(xb, 0, DIM, wvb, 0, DIM, vtb, 0, 0,   bv, DIM);

    // 3) scores = q . k^T, per batch: M = Nc = SEQ, K = DIM
    dim3 gs(SEQ / 256, SEQ / 128, BATCH);
    wmma_gemm<0><<<gs, 256, 0, stream>>>(qb, (size_t)SEQ * DIM, DIM,
                                         kb, (size_t)SEQ * DIM, DIM,
                                         sc, (size_t)SEQ * SEQ, SEQ, nullptr, DIM);

    // 4) softmax rows (in place, bf16)
    softmax_rows<<<BATCH * SEQ, 256, 0, stream>>>(sc);

    // 5) out = probs . v, per batch: M = SEQ, Nc = DIM, K = SEQ
    dim3 go(DIM / 256, SEQ / 128, BATCH);
    wmma_gemm<2><<<go, 256, 0, stream>>>(sc,  (size_t)SEQ * SEQ, SEQ,
                                         vtb, (size_t)DIM * SEQ, SEQ,
                                         (float*)d_out, (size_t)SEQ * DIM, DIM,
                                         nullptr, SEQ);
}
